// TemporalAttention_23948737643066
// MI455X (gfx1250) — matrix operand order repaired and checked by reading, not compiled, not run on hardware
//
#include <hip/hip_runtime.h>
#include <hip/hip_bf16.h>

typedef __attribute__((ext_vector_type(16))) __bf16 v16bf;
typedef __attribute__((ext_vector_type(8)))  float  v8f;

#define NNBR    32
#define DFEAT   128
#define OUTD    256
#define KEYD    384
#define KVOUT   512
#define NHEADS  8
#define HDIM    32
#define GB      2          // batch elements per block (L2 reuse factor)

// padded LDS strides (elements)
#define SX_LD   392        // 384 + 8 bf16
#define SKV_LD  520        // 512 + 8 bf16

union Frag {
  v16bf  v;
  float4 f4[2];
};

// ---------------- prep: W_KV [384,512] f32  ->  W_KV^T [512,384] bf16 ----------------
__global__ void __launch_bounds__(256)
prep_wkvT(const float* __restrict__ wkv, __hip_bfloat16* __restrict__ wkvT) {
  int idx = blockIdx.x * 256 + threadIdx.x;
  if (idx < KVOUT * KEYD) {
    int o = idx / KEYD;            // output column 0..511 (row of transposed)
    int k = idx - o * KEYD;        // 0..383
    wkvT[idx] = __float2bfloat16(wkv[k * KVOUT + o]);
  }
}

// ------------- fused temporal attention, GB batch elements per block -------------
__global__ void __launch_bounds__(256)
temporal_attn(const float* __restrict__ node_feat,
              const float* __restrict__ time_feat,
              const float* __restrict__ edge_feat,
              const float* __restrict__ nbr_node_feat,
              const float* __restrict__ nbr_time_feat,
              const int*   __restrict__ nbr_mask,
              const float* __restrict__ W_Q,
              const __hip_bfloat16* __restrict__ WkvT,
              const float* __restrict__ W_O,
              const float* __restrict__ b_O,
              const float* __restrict__ ln_gamma,
              const float* __restrict__ ln_beta,
              float* __restrict__ out)
{
  // Union region: X tiles (GEMM phase) overlaid by KV tile (attention phase).
  // sizeof(sX) = GB*32*392*2 = 50176 B >= sizeof(sKV) = 32*520*2 = 33280 B
  __shared__ __align__(16) unsigned char sMem[GB * NNBR * SX_LD * 2];
  __hip_bfloat16 (*sX)[NNBR][SX_LD] = (__hip_bfloat16 (*)[NNBR][SX_LD])sMem;
  __hip_bfloat16 (*sKV)[SKV_LD]     = (__hip_bfloat16 (*)[SKV_LD])sMem;

  __shared__ float sR[GB][OUTD];    // residuals (f32)
  __shared__ float sQ[GB][OUTD];
  __shared__ float sA[NHEADS][NNBR];
  __shared__ float sO[OUTD];
  __shared__ float sRed[16];

  const int b0   = blockIdx.x * GB;
  const int tid  = threadIdx.x;
  const int lane = tid & 31;
  const int w    = tid >> 5;

  // ---- Stage 1: residuals R and X tiles (f32 -> bf16) ----
  #pragma unroll
  for (int g = 0; g < GB; ++g) {
    const int b = b0 + g;
    sR[g][tid] = (tid < DFEAT) ? node_feat[b * DFEAT + tid]
                               : time_feat[b * DFEAT + (tid - DFEAT)];
  }
  // X: [GB][32][384]; cols 0..127 nbr_node, 128..255 edge, 256..383 nbr_time
  #pragma unroll
  for (int it = 0; it < GB * 12; ++it) {
    int f4 = it * 256 + tid;             // float4 slots over [GB][32][96]
    int g  = f4 / 3072;
    int r  = f4 - g * 3072;
    int n  = r / 96;
    int c4 = r - n * 96;                 // 0..95
    const float* src;
    int coff;
    if (c4 < 32)      { src = nbr_node_feat; coff = c4 * 4; }
    else if (c4 < 64) { src = edge_feat;     coff = (c4 - 32) * 4; }
    else              { src = nbr_time_feat; coff = (c4 - 64) * 4; }
    float4 v = *(const float4*)&src[((size_t)(b0 + g) * NNBR + n) * DFEAT + coff];
    int c = c4 * 4;
    sX[g][n][c + 0] = __float2bfloat16(v.x);
    sX[g][n][c + 1] = __float2bfloat16(v.y);
    sX[g][n][c + 2] = __float2bfloat16(v.z);
    sX[g][n][c + 3] = __float2bfloat16(v.w);
  }
  __syncthreads();

  // ---- Stage 2: Q = R @ W_Q per batch (GEMV, f32 VALU) ----
  #pragma unroll
  for (int g = 0; g < GB; ++g) {
    float q = 0.f;
    #pragma unroll 4
    for (int i = 0; i < OUTD; ++i)
      q += sR[g][i] * W_Q[i * OUTD + tid];
    sQ[g][tid] = q;
  }

  // ---- Stage 3: KV[g][32,512] = X[g] @ W_KV via bf16 WMMA; B frag reused GB*2 times ----
  v8f acc[GB][2][4];
  #pragma unroll
  for (int g = 0; g < GB; ++g)
    #pragma unroll
    for (int i = 0; i < 2; ++i)
      #pragma unroll
      for (int j = 0; j < 4; ++j)
        acc[g][i][j] = (v8f){0.f, 0.f, 0.f, 0.f, 0.f, 0.f, 0.f, 0.f};

  const int ncolbase = w * 64;        // this wave's first output column
  const int arow = lane & 15;
  const int asel = lane >> 4;         // half-wave select

  #pragma unroll
  for (int kt = 0; kt < 12; ++kt) {
    // A fragments (ISA 16-bit A layout): lane<16 -> K {0..7}+{16..23}; lane>=16 -> {8..15}+{24..31}
    Frag a[GB][2];
    const int cb = kt * 32 + asel * 8;
    #pragma unroll
    for (int g = 0; g < GB; ++g) {
      a[g][0].f4[0] = *(const float4*)&sX[g][arow][cb];
      a[g][0].f4[1] = *(const float4*)&sX[g][arow][cb + 16];
      a[g][1].f4[0] = *(const float4*)&sX[g][arow + 16][cb];
      a[g][1].f4[1] = *(const float4*)&sX[g][arow + 16][cb + 16];
    }
    #pragma unroll
    for (int nt = 0; nt < 4; ++nt) {
      // B fragment: lane -> col N=lane%16, K {0..7}+{16..23} (lane<16) / {8..15}+{24..31} (lane>=16), matching A
      Frag bf;
      const __hip_bfloat16* pB =
          WkvT + (size_t)(ncolbase + nt * 16 + arow) * KEYD + kt * 32 + asel * 8;
      bf.f4[0] = *(const float4*)pB;
      bf.f4[1] = *(const float4*)(pB + 16);
      #pragma unroll
      for (int g = 0; g < GB; ++g) {
        acc[g][0][nt] = __builtin_amdgcn_wmma_f32_16x16x32_bf16(
            false, a[g][0].v, false, bf.v, (short)0, acc[g][0][nt], false, false);
        acc[g][1][nt] = __builtin_amdgcn_wmma_f32_16x16x32_bf16(
            false, a[g][1].v, false, bf.v, (short)0, acc[g][1][nt], false, false);
      }
    }
  }

  // ---- Stage 4: per-batch attention + epilogue (sX region recycled as sKV) ----
  #pragma unroll
  for (int g = 0; g < GB; ++g) {
    const int b = b0 + g;
    __syncthreads();   // all sX reads (g==0) / previous-batch LDS use complete

    // write K|V of batch g to LDS. D layout: lane -> col N=lane%16, vgpr r -> row r + 8*(lane>>4)
    {
      const int mh = asel * 8;
      const int nl = lane & 15;
      #pragma unroll
      for (int mt = 0; mt < 2; ++mt)
        #pragma unroll
        for (int nt = 0; nt < 4; ++nt) {
          const int col = ncolbase + nt * 16 + nl;
          #pragma unroll
          for (int r = 0; r < 8; ++r)
            sKV[mt * 16 + mh + r][col] = __float2bfloat16(acc[g][mt][nt][r]);
        }
    }
    __syncthreads();

    // scores + masked softmax (wave w == head h, lane == neighbor n)
    {
      const int h = w, n = lane;
      float s = 0.f;
      #pragma unroll
      for (int d = 0; d < HDIM; ++d)
        s += sQ[g][h * HDIM + d] * __bfloat162float(sKV[n][h * HDIM + d]);
      s *= 0.17677669529663687f;                  // 1/sqrt(32)
      if (nbr_mask[b * NNBR + n] == 0) s = -1e10f;

      float m = s;
      #pragma unroll
      for (int off = 16; off > 0; off >>= 1)
        m = fmaxf(m, __shfl_xor(m, off, 32));
      float e = __expf(s - m);
      float sum = e;
      #pragma unroll
      for (int off = 16; off > 0; off >>= 1)
        sum += __shfl_xor(sum, off, 32);
      sA[h][n] = e / sum;
    }

    // O[h,d] = sum_n A[h,n] * V[n,h,d]   (sA produced by same wave)
    {
      const int h = w, d = lane;
      float o = 0.f;
      #pragma unroll
      for (int n = 0; n < NNBR; ++n)
        o += sA[h][n] * __bfloat162float(sKV[n][OUTD + h * HDIM + d]);
      sO[h * HDIM + d] = o;
    }
    __syncthreads();

    // out = O @ W_O + b_O ; residual ; LayerNorm
    {
      float o = b_O[tid];
      #pragma unroll 4
      for (int i = 0; i < OUTD; ++i)
        o += sO[i] * W_O[i * OUTD + tid];
      float x = o + sR[g][tid];

      float s1 = x, s2 = x * x;
      #pragma unroll
      for (int off = 16; off > 0; off >>= 1) {
        s1 += __shfl_xor(s1, off, 32);
        s2 += __shfl_xor(s2, off, 32);
      }
      if (lane == 0) { sRed[w] = s1; sRed[8 + w] = s2; }
      __syncthreads();
      float S1 = 0.f, S2 = 0.f;
      #pragma unroll
      for (int i = 0; i < 8; ++i) { S1 += sRed[i]; S2 += sRed[8 + i]; }

      const float mu  = S1 * (1.f / OUTD);
      const float var = S2 * (1.f / OUTD) - mu * mu;
      const float y = (x - mu) * rsqrtf(var + 1e-5f) * ln_gamma[tid] + ln_beta[tid];
      out[(size_t)b * OUTD + tid] = y;
    }
  }
}

// ---------------- host launcher ----------------
extern "C" void kernel_launch(void* const* d_in, const int* in_sizes, int n_in,
                              void* d_out, int out_size, void* d_ws, size_t ws_size,
                              hipStream_t stream) {
  const float* node_feat = (const float*)d_in[0];
  const float* time_feat = (const float*)d_in[1];
  const float* edge_feat = (const float*)d_in[2];
  const float* nbr_node  = (const float*)d_in[3];
  const float* nbr_time  = (const float*)d_in[4];
  const int*   nbr_mask  = (const int*)d_in[5];
  const float* W_Q  = (const float*)d_in[6];
  const float* W_KV = (const float*)d_in[7];
  const float* W_O  = (const float*)d_in[8];
  const float* b_O  = (const float*)d_in[9];
  const float* gam  = (const float*)d_in[10];
  const float* bet  = (const float*)d_in[11];

  // workspace: W_KV^T as bf16, 512*384*2 = 393216 bytes
  __hip_bfloat16* WkvT = (__hip_bfloat16*)d_ws;

  const int B = in_sizes[0] / DFEAT;   // 8192

  prep_wkvT<<<(KVOUT * KEYD + 255) / 256, 256, 0, stream>>>(W_KV, WkvT);
  temporal_attn<<<B / GB, 256, 0, stream>>>(node_feat, time_feat, edge_feat,
                                            nbr_node, nbr_time, nbr_mask,
                                            W_Q, WkvT, W_O, b_O, gam, bet,
                                            (float*)d_out);
}
